// LocalWindowSelfAttention_35201551958426
// MI455X (gfx1250) — matrix operand order repaired and checked by reading, not compiled, not run on hardware
//
#include <hip/hip_runtime.h>

#define WWIN 7
#define NTOK 49
#define MPAD 64
#define CH 256
#define NH 8
#define DH 32
#define IMG 224
#define NPIX (IMG * IMG)
#define QKV_N 768

typedef __bf16 bf16_t;
typedef bf16_t bf16x16 __attribute__((ext_vector_type(16)));
typedef float f32x8 __attribute__((ext_vector_type(8)));
typedef unsigned int u32x8 __attribute__((ext_vector_type(8)));

__device__ __forceinline__ unsigned short f2bf(float f) {
  unsigned int u = __float_as_uint(f);
  u += 0x7FFFu + ((u >> 16) & 1u);   // round-to-nearest-even
  return (unsigned short)(u >> 16);
}

__device__ __forceinline__ bf16x16 pack8(uint4 lo, uint4 hi) {
  u32x8 w;
  w[0] = lo.x; w[1] = lo.y; w[2] = lo.z; w[3] = lo.w;
  w[4] = hi.x; w[5] = hi.y; w[6] = hi.z; w[7] = hi.w;
  return __builtin_bit_cast(bf16x16, w);
}

// A fragment (16x32 bf16, MxK) from row-major [m][k] with row stride ld.
// Layout: lanes 0-15 hold K 0..15, lanes 16-31 hold K 16..31; 2 consecutive K per VGPR.
__device__ __forceinline__ bf16x16 load_A(const unsigned short* base, int ld,
                                          int mBase, int kBase, int lane) {
  int m = mBase + (lane & 15);
  int kh = (lane & 16) ? 16 : 0;
  const unsigned short* p = base + m * ld + kBase + kh;
  uint4 lo = *(const uint4*)(p);
  uint4 hi = *(const uint4*)(p + 8);
  return pack8(lo, hi);
}

// B fragment (32x16 bf16, KxN) from row-major W[n][k] with row stride ld.
// Layout (contiguous k order, same as load_A): lanes 0-15 hold K 0..15,
// lanes 16-31 hold K 16..31; 2 consecutive K per VGPR.
__device__ __forceinline__ bf16x16 load_B_nk(const unsigned short* base, int ld,
                                             int nBase, int kBase, int lane) {
  int n = nBase + (lane & 15);
  int kh = (lane & 16) ? 16 : 0;
  const unsigned short* p = base + n * ld + kBase + kh;
  uint4 lo = *(const uint4*)(p);
  uint4 hi = *(const uint4*)(p + 8);
  return pack8(lo, hi);
}

// B fragment (32x16, KxN) from k-major storage V[k][n] with row stride ld.
__device__ __forceinline__ bf16x16 load_B_kn(const unsigned short* base, int ld,
                                             int nBase, int kBase, int lane) {
  int n = nBase + (lane & 15);
  int kb = kBase + ((lane & 16) ? 16 : 0);
  u32x8 w;
#pragma unroll
  for (int v = 0; v < 8; ++v) {
    unsigned e0 = base[(kb + 2 * v) * ld + n];
    unsigned e1 = base[(kb + 2 * v + 1) * ld + n];
    w[v] = e0 | (e1 << 16);
  }
  return __builtin_bit_cast(bf16x16, w);
}

__device__ __forceinline__ f32x8 wmma_bf16(bf16x16 a, bf16x16 b, f32x8 c) {
  return __builtin_amdgcn_wmma_f32_16x16x32_bf16(false, a, false, b, (short)0, c,
                                                 false, false);
}

// ---------- weight fp32 -> bf16 pre-conversion (weights then live in L2) ----
__global__ __launch_bounds__(256) void cvt_weights(
    const float* __restrict__ wq, const float* __restrict__ wp,
    unsigned short* __restrict__ dst) {
  int i = blockIdx.x * 256 + threadIdx.x;
  if (i < QKV_N * CH) dst[i] = f2bf(wq[i]);
  if (i < CH * CH) dst[QKV_N * CH + i] = f2bf(wp[i]);
}

// ---------- fused per-window attention: 1 block = 1 (7x7) window ------------
__global__ __launch_bounds__(256) void swin_window_attn(
    const float* __restrict__ x, const float* __restrict__ qkv_b,
    const float* __restrict__ proj_b, const float* __restrict__ rpb_table,
    const unsigned short* __restrict__ wqkv,
    const unsigned short* __restrict__ wproj, float* __restrict__ out) {
  extern __shared__ char smem_raw[];
  unsigned short* lds_x = (unsigned short*)smem_raw;       // 64x256 bf16 (reused as attn-out)
  unsigned short* lds_qkv = lds_x + MPAD * CH;             // 3 x 64x256 bf16
  unsigned short* lds_P = lds_qkv + 3 * MPAD * CH;         // 8 waves x 16x64 bf16
  float* lds_rpb = (float*)(lds_P + NH * 16 * 64);         // 169 x 8 f32

  const int tid = threadIdx.x;
  const int lane = tid & 31;
  const int wave = tid >> 5;
  const int nCol = lane & 15;
  const int rOff = (lane & 16) ? 8 : 0;

  const int win = blockIdx.x;
  const int b = win >> 10;                 // 1024 windows per batch image
  const int wh0 = ((win >> 5) & 31) * WWIN;
  const int ww0 = (win & 31) * WWIN;

  for (int i = tid; i < 169 * NH; i += 256) lds_rpb[i] = rpb_table[i];

  // load 49 window tokens -> LDS bf16, zero-pad rows 49..63
  for (int i = tid; i < MPAD * CH; i += 256) {
    int row = i >> 8, ch = i & 255;
    float v = 0.f;
    if (row < NTOK) {
      int pr = wh0 + row / WWIN, pc = ww0 + row % WWIN;
      v = x[((size_t)b * NPIX + (size_t)pr * IMG + pc) * CH + ch];
    }
    lds_x[i] = f2bf(v);
  }
  __syncthreads();

  // ---- stage 1: QKV GEMM (64x256)@(256x768), 6 N-tiles per wave ----
  const float scale = 0.17677669529663687f;  // 32^-0.5
  for (int nt = wave * 6; nt < wave * 6 + 6; ++nt) {
    int nBase = nt * 16;
    f32x8 acc[4] = {};
#pragma unroll 2
    for (int ks = 0; ks < CH; ks += 32) {
      bf16x16 bfrag = load_B_nk(wqkv, CH, nBase, ks, lane);
#pragma unroll
      for (int mt = 0; mt < 4; ++mt) {
        bf16x16 afrag = load_A(lds_x, CH, mt * 16, ks, lane);
        acc[mt] = wmma_bf16(afrag, bfrag, acc[mt]);
      }
    }
    int col = nBase + nCol;
    float bias = qkv_b[col];
    float scl = (col < CH) ? scale : 1.f;  // scale only q columns
    unsigned short* dst = lds_qkv + (col >> 8) * (MPAD * CH) + (col & 255);
#pragma unroll
    for (int mt = 0; mt < 4; ++mt)
#pragma unroll
      for (int i = 0; i < 8; ++i) {
        int m = mt * 16 + i + rOff;
        dst[m * CH] = f2bf((acc[mt][i] + bias) * scl);
      }
  }
  __syncthreads();

  // ---- stage 2: attention, one head per wave ----
  const int head = wave;
  const unsigned short* q_l = lds_qkv;
  const unsigned short* k_l = lds_qkv + MPAD * CH;
  const unsigned short* v_l = lds_qkv + 2 * MPAD * CH;
  unsigned short* P = lds_P + wave * 16 * 64;

  for (int mt = 0; mt < 4; ++mt) {
    bf16x16 aq = load_A(q_l, CH, mt * 16, head * DH, lane);
    f32x8 s[4];
#pragma unroll
    for (int nt = 0; nt < 4; ++nt) {
      bf16x16 bk = load_B_nk(k_l, CH, nt * 16, head * DH, lane);
      f32x8 z = {};
      s[nt] = wmma_bf16(aq, bk, z);
    }
    // relative-position bias + key-padding mask
#pragma unroll
    for (int nt = 0; nt < 4; ++nt) {
      int n = nt * 16 + nCol;
#pragma unroll
      for (int i = 0; i < 8; ++i) {
        int m = mt * 16 + i + rOff;
        if (n < NTOK && m < NTOK) {
          int ridx = (m / WWIN - n / WWIN + WWIN - 1) * (2 * WWIN - 1) +
                     (m % WWIN - n % WWIN + WWIN - 1);
          s[nt][i] += lds_rpb[ridx * NH + head];
        } else if (n >= NTOK) {
          s[nt][i] = -1e30f;
        }
      }
    }
    // row-wise softmax (row = i + rOff; 16 lanes of a half-wave share a row)
#pragma unroll
    for (int i = 0; i < 8; ++i) {
      float mx = fmaxf(fmaxf(s[0][i], s[1][i]), fmaxf(s[2][i], s[3][i]));
#pragma unroll
      for (int off = 1; off < 16; off <<= 1) mx = fmaxf(mx, __shfl_xor(mx, off, 32));
      float sum = 0.f;
#pragma unroll
      for (int nt = 0; nt < 4; ++nt) {
        float e = __expf(s[nt][i] - mx);
        s[nt][i] = e;
        sum += e;
      }
#pragma unroll
      for (int off = 1; off < 16; off <<= 1) sum += __shfl_xor(sum, off, 32);
      float inv = 1.f / sum;
#pragma unroll
      for (int nt = 0; nt < 4; ++nt) s[nt][i] *= inv;
    }
    // P (16x64) -> per-wave LDS scratch for A-fragment re-layout
#pragma unroll
    for (int nt = 0; nt < 4; ++nt) {
      int n = nt * 16 + nCol;
#pragma unroll
      for (int i = 0; i < 8; ++i) P[(i + rOff) * 64 + n] = f2bf(s[nt][i]);
    }
    asm volatile("s_wait_dscnt 0" ::: "memory");
    // O = P(16x64) @ V(64x32)
    f32x8 o0 = {}, o1 = {};
#pragma unroll
    for (int kt = 0; kt < 2; ++kt) {
      bf16x16 ap = load_A(P, 64, 0, kt * 32, lane);
      bf16x16 bv0 = load_B_kn(v_l + head * DH, CH, 0, kt * 32, lane);
      bf16x16 bv1 = load_B_kn(v_l + head * DH, CH, 16, kt * 32, lane);
      o0 = wmma_bf16(ap, bv0, o0);
      o1 = wmma_bf16(ap, bv1, o1);
    }
#pragma unroll
    for (int i = 0; i < 8; ++i) {
      int m = mt * 16 + i + rOff;
      lds_x[m * CH + head * DH + nCol] = f2bf(o0[i]);
      lds_x[m * CH + head * DH + 16 + nCol] = f2bf(o1[i]);
    }
  }
  __syncthreads();

  // ---- stage 3: proj GEMM (64x256)@(256x256) + bias, store fp32 ----
  for (int nt = wave * 2; nt < wave * 2 + 2; ++nt) {
    int nBase = nt * 16;
    f32x8 acc[4] = {};
#pragma unroll 2
    for (int ks = 0; ks < CH; ks += 32) {
      bf16x16 bfrag = load_B_nk(wproj, CH, nBase, ks, lane);
#pragma unroll
      for (int mt = 0; mt < 4; ++mt) {
        bf16x16 afrag = load_A(lds_x, CH, mt * 16, ks, lane);
        acc[mt] = wmma_bf16(afrag, bfrag, acc[mt]);
      }
    }
    int col = nBase + nCol;
    float pb = proj_b[col];
#pragma unroll
    for (int mt = 0; mt < 4; ++mt)
#pragma unroll
      for (int i = 0; i < 8; ++i) {
        int m = mt * 16 + i + rOff;
        if (m < NTOK) {
          int pr = wh0 + m / WWIN, pc = ww0 + m % WWIN;
          out[((size_t)b * NPIX + (size_t)pr * IMG + pc) * CH + col] =
              acc[mt][i] + pb;
        }
      }
  }
}

extern "C" void kernel_launch(void* const* d_in, const int* in_sizes, int n_in,
                              void* d_out, int out_size, void* d_ws,
                              size_t ws_size, hipStream_t stream) {
  const float* x = (const float*)d_in[0];
  const float* qkv_w = (const float*)d_in[1];
  const float* qkv_b = (const float*)d_in[2];
  const float* proj_w = (const float*)d_in[3];
  const float* proj_b = (const float*)d_in[4];
  const float* rpb = (const float*)d_in[5];
  // d_in[6]=H, d_in[7]=W are fixed at 224 by the harness setup (no padding case)
  unsigned short* wbf = (unsigned short*)d_ws;
  float* out = (float*)d_out;

  cvt_weights<<<QKV_N, 256, 0, stream>>>(qkv_w, proj_w, wbf);

  size_t smem = (size_t)(MPAD * CH + 3 * MPAD * CH + NH * 16 * 64) * 2 +
                (size_t)169 * NH * 4;  // ~150 KB of the 320 KB WGP LDS
  swin_window_attn<<<8 * 32 * 32, 256, smem, stream>>>(
      x, qkv_b, proj_b, rpb, wbf, wbf + QKV_N * CH, out);
}
